// SwinBlock_16080357556465
// MI455X (gfx1250) — compile-verified, matrix operand order checked by reading, not run on hardware
//
#include <hip/hip_runtime.h>
#include <hip/hip_bf16.h>
#include <cstdint>

typedef __attribute__((ext_vector_type(16))) _Float16 v16h;
typedef __attribute__((ext_vector_type(8)))  float    v8f;

#define WMMA_F16(a, b, c) \
  __builtin_amdgcn_wmma_f32_16x16x32_f16(false, (a), false, (b), (short)0, (c), false, false)

// Async global -> LDS copy of 16 bytes per lane (CDNA5 TDM-adjacent path,
// tracked by ASYNCcnt).  lds_off = LDS byte offset (low 32 bits of a generic
// LDS pointer), gptr = per-lane 64-bit global address.
__device__ __forceinline__ void async_ld_b128(uint32_t lds_off, const void* gptr) {
  asm volatile("global_load_async_to_lds_b128 %0, %1, off"
               :: "v"(lds_off), "v"((uint64_t)(uintptr_t)gptr)
               : "memory");
}
#define WAIT_ASYNC(n) asm volatile("s_wait_asynccnt %0" :: "n"(n) : "memory")

// Load a 16x32 f16 A-operand (or B-operand as A-layout of B^T) from a
// row-major tile. CDNA5 layout: lanes 0-15 hold M=lane, halves 0..7 -> K=kh+i,
// halves 8..15 -> K=16+kh+i with kh = (lane>>4)*8.
__device__ __forceinline__ v16h lds_load_a(const _Float16* base, int ld, int lane) {
  int m = lane & 15, kh = (lane >> 4) << 3;
  const _Float16* p = base + m * ld + kh;
  v16h r;
#pragma unroll
  for (int i = 0; i < 8; ++i) r[i] = p[i];
#pragma unroll
  for (int i = 0; i < 8; ++i) r[8 + i] = p[16 + i];
  return r;
}

// B-operand for a KxN tile stored row-major with leading dim ld:
// element (n, k) = base[k*ld + n].
__device__ __forceinline__ v16h load_b_kxn(const _Float16* base, int ld, int lane) {
  int n = lane & 15, kh = (lane >> 4) << 3;
  v16h r;
#pragma unroll
  for (int i = 0; i < 8; ++i) r[i] = base[(kh + i) * ld + n];
#pragma unroll
  for (int i = 0; i < 8; ++i) r[8 + i] = base[(16 + kh + i) * ld + n];
  return r;
}

__global__ void cvt_f32_to_f16(const float* __restrict__ in, _Float16* __restrict__ out, int n) {
  int i = blockIdx.x * blockDim.x + threadIdx.x;
  if (i < n) out[i] = (_Float16)in[i];
}

// Weight pre-pass: fp32 [K,N] -> f16 transposed [N,K] (done once; GEMM B tiles
// then stage with a single contiguous async b128 per thread).
__global__ void cvt_w_t(const float* __restrict__ src, _Float16* __restrict__ dst,
                        int K, int N) {
  int i = blockIdx.x * blockDim.x + threadIdx.x;
  if (i < K * N) {
    int k = i / N, n = i - k * N;
    dst[(size_t)n * K + k] = (_Float16)src[i];
  }
}

// ---------------------------------------------------------------------------
// WMMA GEMM: C[M,N] = A_f16[M,K] @ B (given as BtG f16 [N,K]) + bias.
// Block tile 128x64, 8 waves, wave tile 32x32 (2x2 WMMA), K-step 32.
// Double-buffered LDS staging via global_load_async_to_lds_b128:
// 3 async ops per wave per tile; s_wait_asynccnt 3 => previous tile landed.
// Epilogues: 0 f16 store (QKV) | 1 window-reverse+roll scatter + residual
//            2 exact GELU f16 (MLP1)  | 3 +x1 residual f32 (MLP2 -> out)
// ---------------------------------------------------------------------------
template <int EPI>
__global__ void __launch_bounds__(256) gemm_kernel(
    const _Float16* __restrict__ A, const _Float16* __restrict__ BtG,
    const float* __restrict__ bias, const float* __restrict__ resid,
    float* __restrict__ outF, _Float16* __restrict__ outH,
    int M, int N, int K) {
  __shared__ _Float16 As[2][128][40];   // padded ld=40 -> conflict-free b128 reads
  __shared__ _Float16 Bs[2][64][40];    // B tile transposed: Bs[n][k]
  int tid  = threadIdx.x;
  int wave = tid >> 5, lane = tid & 31;
  int wm = wave & 3, wn = wave >> 2;
  int mBase = blockIdx.x * 128, nBase = blockIdx.y * 64;
  v8f acc[2][2] = {};
  int ar = tid >> 1, ac = (tid & 1) << 4;   // A stage: 32B (2 x b128) per thread
  int bn = tid >> 2, bseg = (tid & 3) << 3; // B stage: 16B (1 x b128) per thread

  const _Float16* gA = A + (size_t)(mBase + ar) * K + ac;
  const _Float16* gB = BtG + (size_t)(nBase + bn) * K + bseg;

  // prefetch tile 0 -> buffer 0
  async_ld_b128((uint32_t)(uintptr_t)&As[0][ar][ac],     gA);
  async_ld_b128((uint32_t)(uintptr_t)&As[0][ar][ac + 8], gA + 8);
  async_ld_b128((uint32_t)(uintptr_t)&Bs[0][bn][bseg],   gB);

  int ktCount = K >> 5;
  int buf = 0;
  for (int kt = 0; kt < ktCount; ++kt) {
    if (kt + 1 < ktCount) {
      int kn = (kt + 1) << 5;
      async_ld_b128((uint32_t)(uintptr_t)&As[buf ^ 1][ar][ac],     gA + kn);
      async_ld_b128((uint32_t)(uintptr_t)&As[buf ^ 1][ar][ac + 8], gA + kn + 8);
      async_ld_b128((uint32_t)(uintptr_t)&Bs[buf ^ 1][bn][bseg],   gB + kn);
      WAIT_ASYNC(3);          // tile kt complete (async loads retire in order)
    } else {
      WAIT_ASYNC(0);
    }
    __syncthreads();          // tile kt visible from all waves
    v16h a0 = lds_load_a(&As[buf][wm * 32][0],      40, lane);
    v16h a1 = lds_load_a(&As[buf][wm * 32 + 16][0], 40, lane);
    v16h b0 = lds_load_a(&Bs[buf][wn * 32][0],      40, lane);
    v16h b1 = lds_load_a(&Bs[buf][wn * 32 + 16][0], 40, lane);
    acc[0][0] = WMMA_F16(a0, b0, acc[0][0]);
    acc[0][1] = WMMA_F16(a0, b1, acc[0][1]);
    acc[1][0] = WMMA_F16(a1, b0, acc[1][0]);
    acc[1][1] = WMMA_F16(a1, b1, acc[1][1]);
    __syncthreads();          // done reading buf before tile kt+2 overwrites it
    buf ^= 1;
  }

  int hi = lane >> 4, lo = lane & 15;
#pragma unroll
  for (int i = 0; i < 2; ++i)
#pragma unroll
    for (int j = 0; j < 2; ++j)
#pragma unroll
      for (int r = 0; r < 8; ++r) {
        int row = mBase + wm * 32 + i * 16 + r + 8 * hi;
        int col = nBase + wn * 32 + j * 16 + lo;
        float v = acc[i][j][r] + bias[col];
        if (EPI == 0) {
          outH[(size_t)row * N + col] = (_Float16)v;
        } else if (EPI == 1) {
          // row is window-order: wi*64 + n. Undo shift: scatter to orig coords.
          int wi = row >> 6, n = row & 63;
          int b  = wi >> 8, winh = (wi >> 4) & 15, winw = wi & 15;
          int dh = n >> 3, dw = n & 7;
          int ho = (winh * 8 + dh + 4) & 127;
          int wo = (winw * 8 + dw + 4) & 127;
          size_t orow = (size_t)b * 16384 + ho * 128 + wo;
          outF[orow * 192 + col] = v + resid[orow * 192 + col];
        } else if (EPI == 2) {
          float g = 0.5f * v * (1.0f + erff(v * 0.70710678118654752f));
          outH[(size_t)row * N + col] = (_Float16)g;
        } else {
          outF[(size_t)row * N + col] = v + resid[(size_t)row * N + col];
        }
      }
}

// ---------------------------------------------------------------------------
// Windowed attention: 1 workgroup per window (2048), 6 waves = 1 per head.
// K,V async-staged into LDS; Q tiles loaded from global in A-layout.
// Softmax in f32 with half-wave shuffle reductions; P -> LDS for layout swap.
// ---------------------------------------------------------------------------
__global__ void __launch_bounds__(192) attn_kernel(
    const _Float16* __restrict__ qkvh, const float* __restrict__ rpb,
    _Float16* __restrict__ outh) {
  __shared__ _Float16 sKV[64][384];      // cols 0..191 = K, 192..383 = V
  __shared__ _Float16 sP[6][16][64];     // per-head P row-block scratch
  int wiB = blockIdx.x;
  int b = wiB >> 8, winh = (wiB >> 4) & 15, winw = wiB & 15;
  int tid = threadIdx.x;
  {  // async stage K,V: 64 tokens x 384 halfs; 192 threads x 16 b128 ops
    int tok = tid / 3, seg = tid % 3;
    int dh = tok >> 3, dw = tok & 7;
    int ho = (winh * 8 + dh + 4) & 127;         // roll(-4) gather
    int wo = (winw * 8 + dw + 4) & 127;
    const _Float16* g =
        qkvh + ((size_t)b * 16384 + ho * 128 + wo) * 576 + 192 + seg * 128;
    uint32_t l = (uint32_t)(uintptr_t)&sKV[tok][seg * 128];
#pragma unroll
    for (int i = 0; i < 16; ++i) async_ld_b128(l + i * 16, g + i * 8);
    WAIT_ASYNC(0);
  }
  __syncthreads();

  int head = tid >> 5, lane = tid & 31;
  int hi = lane >> 4, lo = lane & 15;
  const float scale = 0.17677669529663687f;  // 32^-0.5

  for (int ti = 0; ti < 4; ++ti) {
    v16h aq;  // Q tile (16x32) straight from global in A-layout
    {
      int tokq = ti * 16 + lo;
      int kh = hi << 3;
      int dh = tokq >> 3, dw = tokq & 7;
      int ho = (winh * 8 + dh + 4) & 127;
      int wo = (winw * 8 + dw + 4) & 127;
      const _Float16* p =
          qkvh + ((size_t)b * 16384 + ho * 128 + wo) * 576 + head * 32 + kh;
#pragma unroll
      for (int i = 0; i < 8; ++i) aq[i] = p[i];
#pragma unroll
      for (int i = 0; i < 8; ++i) aq[8 + i] = p[16 + i];
    }
    float vals[4][8];
#pragma unroll
    for (int tj = 0; tj < 4; ++tj) {
      v16h bk = lds_load_a(&sKV[tj * 16][head * 32], 384, lane);
      v8f z = {};
      v8f sv = WMMA_F16(aq, bk, z);
#pragma unroll
      for (int r = 0; r < 8; ++r) {
        int iTok = ti * 16 + r + 8 * hi;          // query token
        int jTok = tj * 16 + lo;                  // key token
        int ih = iTok >> 3, iw = iTok & 7, jh = jTok >> 3, jw = jTok & 7;
        float bias = rpb[((ih - jh + 7) * 15 + (iw - jw + 7)) * 6 + head];
        int ha = winh * 8 + ih, wa = winw * 8 + iw;
        int hb = winh * 8 + jh, wb = winw * 8 + jw;
        int regi = (ha < 120 ? 0 : (ha < 124 ? 1 : 2)) * 3 +
                   (wa < 120 ? 0 : (wa < 124 ? 1 : 2));
        int regj = (hb < 120 ? 0 : (hb < 124 ? 1 : 2)) * 3 +
                   (wb < 120 ? 0 : (wb < 124 ? 1 : 2));
        vals[tj][r] = sv[r] * scale + bias + ((regi != regj) ? -100.0f : 0.0f);
      }
    }
    // softmax over 64 keys: 4 tiles x 16 lanes in the same half-wave
#pragma unroll
    for (int r = 0; r < 8; ++r) {
      float mx = fmaxf(fmaxf(vals[0][r], vals[1][r]), fmaxf(vals[2][r], vals[3][r]));
#pragma unroll
      for (int o = 1; o < 16; o <<= 1) mx = fmaxf(mx, __shfl_xor(mx, o, 32));
      float e0 = __expf(vals[0][r] - mx), e1 = __expf(vals[1][r] - mx);
      float e2 = __expf(vals[2][r] - mx), e3 = __expf(vals[3][r] - mx);
      float sum = e0 + e1 + e2 + e3;
#pragma unroll
      for (int o = 1; o < 16; o <<= 1) sum += __shfl_xor(sum, o, 32);
      float inv = 1.0f / sum;
      int rr = r + 8 * hi;  // row inside this 16-row block
      sP[head][rr][0 * 16 + lo] = (_Float16)(e0 * inv);
      sP[head][rr][1 * 16 + lo] = (_Float16)(e1 * inv);
      sP[head][rr][2 * 16 + lo] = (_Float16)(e2 * inv);
      sP[head][rr][3 * 16 + lo] = (_Float16)(e3 * inv);
    }
    // P(16x64) @ V(64x32): 2 output tiles x 2 K-steps
#pragma unroll
    for (int vj = 0; vj < 2; ++vj) {
      v8f acc = {};
#pragma unroll
      for (int ks = 0; ks < 2; ++ks) {
        v16h ap = lds_load_a(&sP[head][0][ks * 32], 64, lane);
        v16h bv = load_b_kxn(&sKV[ks * 32][192 + head * 32 + vj * 16], 384, lane);
        acc = WMMA_F16(ap, bv, acc);
      }
#pragma unroll
      for (int r = 0; r < 8; ++r) {
        int row = ti * 16 + r + 8 * hi;
        outh[((size_t)wiB * 64 + row) * 192 + head * 32 + vj * 16 + lo] =
            (_Float16)acc[r];
      }
    }
  }
}

// LayerNorm over C=192, one wave per row, output f16 for MLP1 A-operand.
__global__ void __launch_bounds__(256) ln_kernel(
    const float* __restrict__ x1, const float* __restrict__ g,
    const float* __restrict__ bta, _Float16* __restrict__ xn) {
  int wave = threadIdx.x >> 5, lane = threadIdx.x & 31;
  size_t row = (size_t)blockIdx.x * 8 + wave;
  const float* p = x1 + row * 192;
  float v[6], s = 0.f, sq = 0.f;
#pragma unroll
  for (int i = 0; i < 6; ++i) {
    v[i] = p[lane + 32 * i];
    s += v[i]; sq += v[i] * v[i];
  }
#pragma unroll
  for (int o = 1; o < 32; o <<= 1) {
    s += __shfl_xor(s, o, 32); sq += __shfl_xor(sq, o, 32);
  }
  float mu = s * (1.0f / 192.0f);
  float var = sq * (1.0f / 192.0f) - mu * mu;
  float inv = rsqrtf(var + 1e-5f);
#pragma unroll
  for (int i = 0; i < 6; ++i) {
    int c = lane + 32 * i;
    xn[row * 192 + c] = (_Float16)((v[i] - mu) * inv * g[c] + bta[c]);
  }
}

extern "C" void kernel_launch(void* const* d_in, const int* in_sizes, int n_in,
                              void* d_out, int out_size, void* d_ws, size_t ws_size,
                              hipStream_t stream) {
  (void)in_sizes; (void)n_in; (void)out_size; (void)ws_size;
  const float* x      = (const float*)d_in[0];
  const float* qkv_w  = (const float*)d_in[1];
  const float* qkv_b  = (const float*)d_in[2];
  const float* proj_w = (const float*)d_in[3];
  const float* proj_b = (const float*)d_in[4];
  const float* rpb    = (const float*)d_in[5];
  const float* n2g    = (const float*)d_in[6];
  const float* n2b    = (const float*)d_in[7];
  const float* w1     = (const float*)d_in[8];
  const float* b1     = (const float*)d_in[9];
  const float* w2     = (const float*)d_in[10];
  const float* b2     = (const float*)d_in[11];
  float* out = (float*)d_out;

  char* ws = (char*)d_ws;
  _Float16* x_h   = (_Float16*)(ws);                 // 50.3 MB (reused as LN out)
  _Float16* qkv_h = (_Float16*)(ws + 50331648ull);   // 151.0 MB
  _Float16* att_h = (_Float16*)(ws + 201326592ull);  // 50.3 MB (window order)
  float*    x1    = (float*)   (ws + 251658240ull);  // 100.7 MB
  _Float16* h_h   = (_Float16*)(ws + 352321536ull);  // 201.3 MB (MLP hidden)
  _Float16* wq_h  = (_Float16*)(ws + 553648128ull);  // f16 weights, [N][K]
  _Float16* wp_h  = (_Float16*)(ws + 554696704ull);
  _Float16* w1_h  = (_Float16*)(ws + 555745280ull);
  _Float16* w2_h  = (_Float16*)(ws + 556793856ull);

  const int M = 131072;  // B * H * W tokens

  cvt_f32_to_f16<<<dim3(M * 192 / 256), 256, 0, stream>>>(x, x_h, M * 192);
  cvt_w_t<<<dim3((192 * 576 + 255) / 256), 256, 0, stream>>>(qkv_w, wq_h, 192, 576);
  cvt_w_t<<<dim3((192 * 192 + 255) / 256), 256, 0, stream>>>(proj_w, wp_h, 192, 192);
  cvt_w_t<<<dim3((192 * 768 + 255) / 256), 256, 0, stream>>>(w1, w1_h, 192, 768);
  cvt_w_t<<<dim3((768 * 192 + 255) / 256), 256, 0, stream>>>(w2, w2_h, 768, 192);

  // QKV: [131072,192] x [192,576]
  gemm_kernel<0><<<dim3(M / 128, 576 / 64), 256, 0, stream>>>(
      x_h, wq_h, qkv_b, nullptr, nullptr, qkv_h, M, 576, 192);

  // Windowed attention (shift folded into gather)
  attn_kernel<<<dim3(2048), 192, 0, stream>>>(qkv_h, rpb, att_h);

  // Proj + window-reverse + roll + residual -> x1
  gemm_kernel<1><<<dim3(M / 128, 192 / 64), 256, 0, stream>>>(
      att_h, wp_h, proj_b, x, x1, nullptr, M, 192, 192);

  // LayerNorm -> f16 (reuse x_h)
  ln_kernel<<<dim3(M / 8), 256, 0, stream>>>(x1, n2g, n2b, x_h);

  // MLP1 + GELU: [131072,192] x [192,768]
  gemm_kernel<2><<<dim3(M / 128, 768 / 64), 256, 0, stream>>>(
      x_h, w1_h, b1, nullptr, nullptr, h_h, M, 768, 192);

  // MLP2 + residual: [131072,768] x [768,192] -> d_out
  gemm_kernel<3><<<dim3(M / 128, 192 / 64), 256, 0, stream>>>(
      h_h, w2_h, b2, x1, out, nullptr, M, 192, 768);
}
